// Model_50440095924321
// MI455X (gfx1250) — matrix operand order repaired and checked by reading, not compiled, not run on hardware
//
#include <hip/hip_runtime.h>
#include <cmath>

// ---------------------------------------------------------------------------
// Model dims (from the reference)
// ---------------------------------------------------------------------------
constexpr int kL  = 4;
constexpr int kE  = 512;
constexpr int kT  = 1024;
constexpr int kV  = 32000;
constexpr int kH  = 8;
constexpr int kD  = 64;
constexpr int kM  = 2048;
constexpr int kHD = kH * kD;   // 512
constexpr int kB  = 2;
constexpr int kBT = kB * kT;   // 2048

// ---------------------------------------------------------------------------
// CDNA5 WMMA types
// ---------------------------------------------------------------------------
typedef __attribute__((ext_vector_type(16))) __bf16 v16bf;
typedef __attribute__((ext_vector_type(8)))  float  v8f;
typedef __attribute__((ext_vector_type(4)))  float  v4f;
typedef __attribute__((ext_vector_type(2)))  unsigned u2;

union BF16Frag { v16bf v; unsigned short u[16]; };

// Force a global (addrspace 1) 128-bit load; generic pointers routed through
// arrays defeat the compiler's address-space inference and fall back to flat.
__device__ __forceinline__ v4f ldg4(const float* p) {
  return *(const __attribute__((address_space(1))) v4f*)p;
}

// fp32 -> bf16 (round-to-nearest) scalar
__device__ __forceinline__ unsigned short f2bf(float f) {
  return (unsigned short)((__float_as_uint(f) + 0x8000u) >> 16);
}
// pack two fp32 into bf16x2 with one v_perm_b32: out = {bf(a) lo, bf(b) hi}
__device__ __forceinline__ unsigned pack_bf16(float a, float b) {
#if __has_builtin(__builtin_amdgcn_perm)
  unsigned ua = __float_as_uint(a) + 0x8000u;
  unsigned ub = __float_as_uint(b) + 0x8000u;
  // {S0=ub (bytes 4-7), S1=ua (bytes 0-3)}; take ua[3:2] then ub[3:2]
  return __builtin_amdgcn_perm(ub, ua, 0x07060302u);
#else
  return (unsigned)f2bf(a) | ((unsigned)f2bf(b) << 16);
#endif
}

// ---------------------------------------------------------------------------
// Block reductions (wave32)
// ---------------------------------------------------------------------------
__device__ __forceinline__ float waveSum(float v) {
  #pragma unroll
  for (int o = 16; o > 0; o >>= 1) v += __shfl_down(v, o, 32);
  return v;
}
__device__ __forceinline__ float waveMax(float v) {
  #pragma unroll
  for (int o = 16; o > 0; o >>= 1) v = fmaxf(v, __shfl_down(v, o, 32));
  return v;
}
__device__ __forceinline__ float blockSum(float v) {
  __shared__ float sm[8];
  int lane = threadIdx.x & 31, w = threadIdx.x >> 5;
  v = waveSum(v);
  __syncthreads();
  if (lane == 0) sm[w] = v;
  __syncthreads();
  float r = 0.f;
  int nw = (blockDim.x + 31) >> 5;
  for (int i = 0; i < nw; ++i) r += sm[i];
  return r;
}
__device__ __forceinline__ float blockMax(float v) {
  __shared__ float sm[8];
  int lane = threadIdx.x & 31, w = threadIdx.x >> 5;
  v = waveMax(v);
  __syncthreads();
  if (lane == 0) sm[w] = v;
  __syncthreads();
  float r = -3.0e38f;
  int nw = (blockDim.x + 31) >> 5;
  for (int i = 0; i < nw; ++i) r = fmaxf(r, sm[i]);
  return r;
}

// ---------------------------------------------------------------------------
// WMMA bf16 GEMM.  C[M,N] = A[M,K] * op(B), fp32 in/out, bf16 compute.
//   TRANS_B = true : B stored [N,K] row-major (C = A B^T), K-contiguous.
//   TRANS_B = false: B stored [K,N] row-major (C = A B),   N-contiguous.
// Batched via blockIdx.z: z -> (zo = z/inner, zi = z%inner).
//
// Block = 256 threads (8 wave32). Block tile 128(M) x 64(N), K stepped by 32.
// Each wave owns a 32x32 patch: 2 A-frags x 2 B-frags -> 4 v_wmma per K-step.
//
// Software-pipelined: the next K-tile is loaded into registers (batched
// global_load_b128 clause) while the current tile's WMMAs issue from LDS;
// fp32->bf16 packing is one v_perm_b32 per pair, stored b64 into LDS.
//
// Requires (true for every call here): M%128==0, N%64==0, K%32==0, and all
// row / batch strides multiples of 4 floats (16B-aligned vectors).
// ---------------------------------------------------------------------------
template <bool TRANS_B>
__global__ __launch_bounds__(256) void gemm_wmma(
    const float* __restrict__ A, int lda, long long sAo, long long sAi,
    const float* __restrict__ B, int ldb, long long sBo, long long sBi,
    float* __restrict__ C, int ldc, long long sCo, long long sCi,
    int Mdim, int Ndim, int Kdim, int inner)
{
  __shared__ alignas(16) unsigned short lds_a[128][40];  // padded bf16 tiles
  __shared__ alignas(16) unsigned short lds_b[64][40];   // normalized [n][k]

  const int tid  = threadIdx.x;
  const int lane = tid & 31;
  const int wave = tid >> 5;

  const int z  = blockIdx.z;
  const int zo = z / inner, zi = z % inner;
  const float* Ab = A + (long long)zo * sAo + (long long)zi * sAi;
  const float* Bb = B + (long long)zo * sBo + (long long)zi * sBi;
  float*       Cb = C + (long long)zo * sCo + (long long)zi * sCi;

  const int bm0 = blockIdx.y * 128;
  const int bn0 = blockIdx.x * 64;

  // wave -> 32x32 patch of the 128x64 block tile
  const int mtile = (wave & 3) * 32;
  const int ntile = (wave >> 2) * 32;

  // ---- hoisted per-thread staging pointers (bumped by a constant per K) ----
  // A tile: 128x32 floats = 1024 float4, 4 per thread.
  const float* apt[4];
  unsigned short* asd[4];
  #pragma unroll
  for (int j = 0; j < 4; ++j) {
    int idx = j * 256 + tid;
    int m = idx >> 3, k4 = (idx & 7) * 4;
    apt[j] = Ab + (size_t)(bm0 + m) * lda + k4;
    asd[j] = &lds_a[m][k4];
  }
  // B tile: 64x32 floats = 512 float4, 2 per thread.
  const float* bpt[2];
  unsigned short* bsd[2];
  #pragma unroll
  for (int j = 0; j < 2; ++j) {
    int idx = j * 256 + tid;
    if (TRANS_B) {
      int n = idx >> 3, k4 = (idx & 7) * 4;
      bpt[j] = Bb + (size_t)(bn0 + n) * ldb + k4;
      bsd[j] = &lds_b[n][k4];
    } else {
      int kk = idx >> 4, n4 = (idx & 15) * 4;
      bpt[j] = Bb + (size_t)kk * ldb + (bn0 + n4);
      bsd[j] = &lds_b[n4][kk];     // writes fan out across 4 rows (stride 40)
    }
  }
  const long long bStep = TRANS_B ? 32 : (long long)32 * ldb;

  v8f c00 = {}, c01 = {}, c10 = {}, c11 = {};
  v4f fa[4], fb[2];

  // ---- prologue: load first K-tile into registers (one load clause) ----
  #pragma unroll
  for (int j = 0; j < 4; ++j) { fa[j] = ldg4(apt[j]); apt[j] += 32; }
  #pragma unroll
  for (int j = 0; j < 2; ++j) { fb[j] = ldg4(bpt[j]); bpt[j] += bStep; }

  for (int k0 = 0; k0 < Kdim; k0 += 32) {
    // ---- convert current tile registers -> bf16 LDS ----
    #pragma unroll
    for (int j = 0; j < 4; ++j) {
      u2 p; p[0] = pack_bf16(fa[j][0], fa[j][1]); p[1] = pack_bf16(fa[j][2], fa[j][3]);
      *(u2*)asd[j] = p;                                   // b64 store, 8B aligned
    }
    #pragma unroll
    for (int j = 0; j < 2; ++j) {
      if (TRANS_B) {
        u2 p; p[0] = pack_bf16(fb[j][0], fb[j][1]); p[1] = pack_bf16(fb[j][2], fb[j][3]);
        *(u2*)bsd[j] = p;
      } else {
        bsd[j][0 * 40] = f2bf(fb[j][0]); bsd[j][1 * 40] = f2bf(fb[j][1]);
        bsd[j][2 * 40] = f2bf(fb[j][2]); bsd[j][3 * 40] = f2bf(fb[j][3]);
      }
    }
    __syncthreads();

    // ---- issue next K-tile loads; they complete under the WMMAs below ----
    if (k0 + 32 < Kdim) {
      #pragma unroll
      for (int j = 0; j < 4; ++j) { fa[j] = ldg4(apt[j]); apt[j] += 32; }
      #pragma unroll
      for (int j = 0; j < 2; ++j) { fb[j] = ldg4(bpt[j]); bpt[j] += bStep; }
      __builtin_prefetch((const void*)apt[0], 0, 1);      // k+2 tile speculative
    }

    // ---- fragments per CDNA5 wave32 16-bit WMMA layouts ----
    // A 16x32: lanes 0-15 K-base 0, lanes 16-31 K-base 8;
    //          halfs 0..7 -> K=kb+j, halfs 8..15 -> K=kb+16+j.
    BF16Frag a0, a1, b0f, b1f;
    const int am0 = mtile + (lane & 15);
    const int am1 = am0 + 16;
    const int akb = (lane >> 4) * 8;
    #pragma unroll
    for (int j = 0; j < 8; ++j) {
      a0.u[j]     = lds_a[am0][akb + j];
      a0.u[j + 8] = lds_a[am0][akb + 16 + j];
      a1.u[j]     = lds_a[am1][akb + j];
      a1.u[j + 8] = lds_a[am1][akb + 16 + j];
    }
    // B 32x16: N = lane&15; same K order as A (K-base 0 / 8, halfs 8..15 -> +16).
    const int bnc0 = ntile + (lane & 15);
    const int bnc1 = bnc0 + 16;
    const int bkb  = (lane >> 4) * 8;
    #pragma unroll
    for (int j = 0; j < 16; ++j) {
      b0f.u[j] = lds_b[bnc0][bkb + (j & 7) + ((j >> 3) << 4)];
      b1f.u[j] = lds_b[bnc1][bkb + (j & 7) + ((j >> 3) << 4)];
    }

    c00 = __builtin_amdgcn_wmma_f32_16x16x32_bf16(false, a0.v, false, b0f.v,
                                                  (short)0, c00, false, false);
    c01 = __builtin_amdgcn_wmma_f32_16x16x32_bf16(false, a0.v, false, b1f.v,
                                                  (short)0, c01, false, false);
    c10 = __builtin_amdgcn_wmma_f32_16x16x32_bf16(false, a1.v, false, b0f.v,
                                                  (short)0, c10, false, false);
    c11 = __builtin_amdgcn_wmma_f32_16x16x32_bf16(false, a1.v, false, b1f.v,
                                                  (short)0, c11, false, false);
    __syncthreads();
  }

  // ---- epilogue: C/D layout — N = lane&15, VGPR i -> M = (lane>>4)*8 + i ----
  const int rm = (lane >> 4) * 8;
  const int cn = lane & 15;
  #pragma unroll
  for (int i = 0; i < 8; ++i) {
    const int gm0 = bm0 + mtile + rm + i;
    const int gm1 = gm0 + 16;
    const int gn0 = bn0 + ntile + cn;
    const int gn1 = gn0 + 16;
    Cb[(size_t)gm0 * ldc + gn0] = c00[i];
    Cb[(size_t)gm0 * ldc + gn1] = c01[i];
    Cb[(size_t)gm1 * ldc + gn0] = c10[i];
    Cb[(size_t)gm1 * ldc + gn1] = c11[i];
  }
  (void)Mdim; (void)Ndim;   // all launches tile exactly
}

// ---------------------------------------------------------------------------
// Elementwise / small kernels
// ---------------------------------------------------------------------------
__global__ void embed_kernel(const int* __restrict__ idx,
                             const float* __restrict__ tok,
                             const float* __restrict__ pos,
                             float* __restrict__ x) {
  int row = blockIdx.x;          // 0..kBT-1
  int t = row % kT;
  int token = idx[row];
  for (int e = threadIdx.x; e < kE; e += blockDim.x)
    x[(size_t)row * kE + e] = tok[(size_t)token * kE + e] + pos[(size_t)t * kE + e];
}

__global__ __launch_bounds__(256) void rms_kernel(const float* __restrict__ x,
                                                  const float* __restrict__ w,
                                                  float* __restrict__ y) {
  int row = blockIdx.x;
  float s = 0.f;
  for (int e = threadIdx.x; e < kE; e += blockDim.x) {
    float v = x[(size_t)row * kE + e];
    s += v * v;
  }
  s = blockSum(s);
  float inv = rsqrtf(s / (float)kE + 1e-5f);
  for (int e = threadIdx.x; e < kE; e += blockDim.x)
    y[(size_t)row * kE + e] = x[(size_t)row * kE + e] * inv * w[e];
}

// causal row softmax in place over scores rows of length kT; row = z*kT + q
__global__ __launch_bounds__(256) void softmax_causal_kernel(float* __restrict__ S,
                                                             float scale) {
  long long row = blockIdx.x;
  int q = (int)(row % kT);
  float* p = S + row * (long long)kT;
  float mx = -3.0e38f;
  for (int k2 = threadIdx.x; k2 <= q; k2 += blockDim.x)
    mx = fmaxf(mx, p[k2] * scale);
  mx = blockMax(mx);
  float sum = 0.f;
  for (int k2 = threadIdx.x; k2 < kT; k2 += blockDim.x) {
    float val = 0.f;
    if (k2 <= q) val = __expf(p[k2] * scale - mx);
    p[k2] = val;
    sum += val;
  }
  sum = blockSum(sum);
  float inv = 1.f / sum;
  for (int k2 = threadIdx.x; k2 < kT; k2 += blockDim.x) p[k2] *= inv;
}

// ra[b,h,t] = (1/sqrt(D)) * sum_e xn[b,t,e] * wr[h,e,t]
__global__ __launch_bounds__(256) void ra_kernel(const float* __restrict__ xn,
                                                 const float* __restrict__ wr_l,
                                                 float* __restrict__ ra,
                                                 float scale) {
  int bh = blockIdx.x;
  int b = bh / kH, h = bh % kH;
  for (int t = threadIdx.x; t < kT; t += blockDim.x) {
    float s = 0.f;
    for (int e = 0; e < kE; ++e)
      s += xn[((size_t)(b * kT + t)) * kE + e] * wr_l[((size_t)(h * kE + e)) * kT + t];
    ra[(size_t)bh * kT + t] = s * scale;
  }
}

// RRPRAM path: scores depend only on k -> online-softmax prefix scan over t.
// One 64-thread block per (b,h); thread d owns dim d.
__global__ __launch_bounds__(64) void rrp_scan_kernel(const float* __restrict__ ra,
                                                      const float* __restrict__ rv,
                                                      float* __restrict__ out) {
  int bh = blockIdx.x;
  int b = bh / kH, h = bh % kH;
  int d = threadIdx.x;
  float acc = 0.f, den = 0.f, m = -3.0e38f;
  for (int t = 0; t < kT; ++t) {
    float s = ra[(size_t)bh * kT + t];
    float mn = fmaxf(m, s);
    float corr = __expf(m - mn);   // 0 on first iteration
    float p = __expf(s - mn);
    acc = acc * corr + p * rv[((size_t)(b * kT + t)) * kHD + h * kD + d];
    den = den * corr + p;
    m = mn;
    out[((size_t)(b * kT + t)) * kHD + h * kD + d] = acc / den;
  }
}

// sc2[b,t] = dot(xn[b,t,:], eb[b,t,:]) / sqrt(E)
__global__ __launch_bounds__(256) void sc2_kernel(const float* __restrict__ xn,
                                                  const float* __restrict__ eb,
                                                  float* __restrict__ sc2,
                                                  float invs) {
  int row = blockIdx.x;
  float s = 0.f;
  for (int e = threadIdx.x; e < kE; e += blockDim.x)
    s += xn[(size_t)row * kE + e] * eb[(size_t)row * kE + e];
  s = blockSum(s);
  if (threadIdx.x == 0) sc2[row] = s * invs;
}

// Janus scores (rank-1: sc2[q]*sc2[k]) + causal softmax, written as P rows.
__global__ __launch_bounds__(256) void ja_softmax_kernel(const float* __restrict__ sc2,
                                                         float* __restrict__ P) {
  int row = blockIdx.x;          // b*kT + q ; P row offset = row*kT
  int b = row / kT, q = row % kT;
  float sq = sc2[row];
  float* p = P + (size_t)row * kT;
  const float* sk = sc2 + (size_t)b * kT;
  float mx = -3.0e38f;
  for (int k2 = threadIdx.x; k2 <= q; k2 += blockDim.x)
    mx = fmaxf(mx, sq * sk[k2]);
  mx = blockMax(mx);
  float sum = 0.f;
  for (int k2 = threadIdx.x; k2 < kT; k2 += blockDim.x) {
    float val = 0.f;
    if (k2 <= q) val = __expf(sq * sk[k2] - mx);
    p[k2] = val;
    sum += val;
  }
  sum = blockSum(sum);
  float inv = 1.f / sum;
  for (int k2 = threadIdx.x; k2 < kT; k2 += blockDim.x) p[k2] *= inv;
}

// Per-head gated combination of the three attention paths.
__global__ void combine_kernel(const float* __restrict__ qkv,
                               const float* __restrict__ rrp,
                               const float* __restrict__ jan,
                               const float* __restrict__ gate_l,  // [H,3]
                               float* __restrict__ out) {
  int row = blockIdx.x;
  for (int f = threadIdx.x; f < kHD; f += blockDim.x) {
    int h = f / kD;
    float g0 = gate_l[h * 3 + 0], g1 = gate_l[h * 3 + 1], g2 = gate_l[h * 3 + 2];
    float mx = fmaxf(g0, fmaxf(g1, g2));
    float e0 = __expf(g0 - mx), e1 = __expf(g1 - mx), e2 = __expf(g2 - mx);
    float inv = 1.f / (e0 + e1 + e2);
    size_t i = (size_t)row * kHD + f;
    out[i] = (e0 * qkv[i] + e1 * rrp[i] + e2 * jan[i]) * inv;
  }
}

__global__ void add_kernel(float* __restrict__ x, const float* __restrict__ y, int n) {
  int i = blockIdx.x * blockDim.x + threadIdx.x;
  if (i < n) x[i] += y[i];
}

__global__ void silu_mul_kernel(float* __restrict__ g, const float* __restrict__ u, int n) {
  int i = blockIdx.x * blockDim.x + threadIdx.x;
  if (i < n) {
    float v = g[i];
    g[i] = (v / (1.f + __expf(-v))) * u[i];
  }
}

// ---------------------------------------------------------------------------
// Host-side GEMM launcher
// ---------------------------------------------------------------------------
static void launch_gemm(hipStream_t st, bool transB,
                        const float* A, int lda, long long sAo, long long sAi,
                        const float* B, int ldb, long long sBo, long long sBi,
                        float* C, int ldc, long long sCo, long long sCi,
                        int M, int N, int K, int batch, int inner) {
  dim3 grid(N / 64, M / 128, batch), blk(256);
  if (transB)
    gemm_wmma<true><<<grid, blk, 0, st>>>(A, lda, sAo, sAi, B, ldb, sBo, sBi,
                                          C, ldc, sCo, sCi, M, N, K, inner);
  else
    gemm_wmma<false><<<grid, blk, 0, st>>>(A, lda, sAo, sAi, B, ldb, sBo, sBi,
                                           C, ldc, sCo, sCi, M, N, K, inner);
}

// ---------------------------------------------------------------------------
// kernel_launch
// ---------------------------------------------------------------------------
extern "C" void kernel_launch(void* const* d_in, const int* in_sizes, int n_in,
                              void* d_out, int out_size, void* d_ws, size_t ws_size,
                              hipStream_t stream) {
  (void)in_sizes; (void)n_in; (void)out_size; (void)ws_size;

  const int*   idx     = (const int*)  d_in[0];
  const float* tok_emb = (const float*)d_in[1];
  const float* pos_emb = (const float*)d_in[2];
  const float* wq      = (const float*)d_in[3];
  const float* wk      = (const float*)d_in[4];
  const float* wv      = (const float*)d_in[5];
  const float* wvr     = (const float*)d_in[6];
  const float* wr      = (const float*)d_in[7];
  const float* wj      = (const float*)d_in[8];
  const float* gate    = (const float*)d_in[9];
  const float* wo      = (const float*)d_in[10];
  const float* rms1_w  = (const float*)d_in[11];
  const float* rms2_w  = (const float*)d_in[12];
  const float* w_gate  = (const float*)d_in[13];
  const float* w_up    = (const float*)d_in[14];
  const float* w_down  = (const float*)d_in[15];
  const float* rms_f_w = (const float*)d_in[16];
  const float* head_w  = (const float*)d_in[17];
  float* out = (float*)d_out;

  // workspace carve-up (fp32); every size is a multiple of 4 floats
  float* ws = (float*)d_ws;
  size_t off = 0;
  auto alloc = [&](size_t n) { float* p = ws + off; off += n; return p; };
  float* x      = alloc((size_t)kBT * kE);
  float* xn     = alloc((size_t)kBT * kE);
  float* qb     = alloc((size_t)kBT * kHD);
  float* kb     = alloc((size_t)kBT * kHD);
  float* vb     = alloc((size_t)kBT * kHD);
  float* rvb    = alloc((size_t)kBT * kHD);
  float* qkvO   = alloc((size_t)kBT * kHD);
  float* rrpO   = alloc((size_t)kBT * kHD);
  float* janO   = alloc((size_t)kBT * kHD);
  float* echo   = alloc((size_t)kBT * kE);
  float* ebuf   = alloc((size_t)kBT * kE);
  float* attc   = alloc((size_t)kBT * kHD);
  float* proj   = alloc((size_t)kBT * kE);
  float* rabuf  = alloc((size_t)kB * kH * kT);
  float* sc2b   = alloc((size_t)kBT);
  float* mlp_g  = alloc((size_t)kBT * kM);
  float* mlp_u  = alloc((size_t)kBT * kM);
  float* scores = alloc((size_t)kB * kH * kT * kT);   // 64 MB, L2-resident

  const float scD = 1.f / sqrtf((float)kD);
  const float scE = 1.f / sqrtf((float)kE);
  const int nXE = kBT * kE, nXM = kBT * kM;

  // embedding
  embed_kernel<<<dim3(kBT), dim3(256), 0, stream>>>(idx, tok_emb, pos_emb, x);

  for (int l = 0; l < kL; ++l) {
    const float* wq_l   = wq     + (size_t)l * kHD * kE;
    const float* wk_l   = wk     + (size_t)l * kHD * kE;
    const float* wv_l   = wv     + (size_t)l * kHD * kE;
    const float* wvr_l  = wvr    + (size_t)l * kHD * kE;
    const float* wr_l   = wr     + (size_t)l * kH * kE * kT;
    const float* wj_l   = wj     + (size_t)l * kE * kE;
    const float* gate_l = gate   + (size_t)l * kH * 3;
    const float* wo_l   = wo     + (size_t)l * kE * kHD;
    const float* r1_l   = rms1_w + (size_t)l * kE;
    const float* r2_l   = rms2_w + (size_t)l * kE;
    const float* wg_l   = w_gate + (size_t)l * kM * kE;
    const float* wu_l   = w_up   + (size_t)l * kM * kE;
    const float* wd_l   = w_down + (size_t)l * kE * kM;

    // pre-attention RMSNorm
    rms_kernel<<<dim3(kBT), dim3(256), 0, stream>>>(x, r1_l, xn);

    // projections: [BT,E] x [HD,E]^T -> [BT,HD]
    launch_gemm(stream, true, xn, kE, 0, 0, wq_l,  kE, 0, 0, qb,  kHD, 0, 0, kBT, kHD, kE, 1, 1);
    launch_gemm(stream, true, xn, kE, 0, 0, wk_l,  kE, 0, 0, kb,  kHD, 0, 0, kBT, kHD, kE, 1, 1);
    launch_gemm(stream, true, xn, kE, 0, 0, wv_l,  kE, 0, 0, vb,  kHD, 0, 0, kBT, kHD, kE, 1, 1);
    launch_gemm(stream, true, xn, kE, 0, 0, wvr_l, kE, 0, 0, rvb, kHD, 0, 0, kBT, kHD, kE, 1, 1);

    // ---- path 1: QK attention ----
    // scores[z=b*H+h] = q @ k^T  (per-head strided batched NT GEMM)
    launch_gemm(stream, true,
                qb, kHD, (long long)kT * kHD, kD,
                kb, kHD, (long long)kT * kHD, kD,
                scores, kT, (long long)kH * kT * kT, (long long)kT * kT,
                kT, kT, kD, kB * kH, kH);
    softmax_causal_kernel<<<dim3(kB * kH * kT), dim3(256), 0, stream>>>(scores, scD);
    // qkv_out = P @ v  (NN batched)
    launch_gemm(stream, false,
                scores, kT, (long long)kH * kT * kT, (long long)kT * kT,
                vb, kHD, (long long)kT * kHD, kD,
                qkvO, kHD, (long long)kT * kHD, kD,
                kT, kD, kT, kB * kH, kH);

    // ---- path 2: RRPRAM (prefix-scan softmax, O(T*D)) ----
    ra_kernel<<<dim3(kB * kH), dim3(256), 0, stream>>>(xn, wr_l, rabuf, scD);
    rrp_scan_kernel<<<dim3(kB * kH), dim3(64), 0, stream>>>(rabuf, rvb, rrpO);

    // ---- path 3: Janus echo-correlation ----
    launch_gemm(stream, true,  xn,   kE, 0, 0, wj_l, kE, 0, 0, echo, kE, 0, 0, kBT, kE, kE, 1, 1);
    launch_gemm(stream, false, echo, kE, 0, 0, wj_l, kE, 0, 0, ebuf, kE, 0, 0, kBT, kE, kE, 1, 1);
    sc2_kernel<<<dim3(kBT), dim3(256), 0, stream>>>(xn, ebuf, sc2b, scE);
    ja_softmax_kernel<<<dim3(kBT), dim3(256), 0, stream>>>(sc2b, scores);  // reuse front of scores
    // jan_out = Pja @ echo (shared across heads; NN batched over b)
    launch_gemm(stream, false,
                scores, kT, (long long)kT * kT, 0,
                echo, kE, (long long)kT * kE, 0,
                janO, kHD, (long long)kT * kHD, 0,
                kT, kHD, kT, kB, 1);

    // gated combine + output projection + residual
    combine_kernel<<<dim3(kBT), dim3(256), 0, stream>>>(qkvO, rrpO, janO, gate_l, attc);
    launch_gemm(stream, true, attc, kHD, 0, 0, wo_l, kHD, 0, 0, proj, kE, 0, 0, kBT, kE, kHD, 1, 1);
    add_kernel<<<dim3(nXE / 256), dim3(256), 0, stream>>>(x, proj, nXE);

    // ---- MLP ----
    rms_kernel<<<dim3(kBT), dim3(256), 0, stream>>>(x, r2_l, xn);
    launch_gemm(stream, true, xn, kE, 0, 0, wg_l, kE, 0, 0, mlp_g, kM, 0, 0, kBT, kM, kE, 1, 1);
    launch_gemm(stream, true, xn, kE, 0, 0, wu_l, kE, 0, 0, mlp_u, kM, 0, 0, kBT, kM, kE, 1, 1);
    silu_mul_kernel<<<dim3(nXM / 256), dim3(256), 0, stream>>>(mlp_g, mlp_u, nXM);
    launch_gemm(stream, true, mlp_g, kM, 0, 0, wd_l, kM, 0, 0, proj, kE, 0, 0, kBT, kE, kM, 1, 1);
    add_kernel<<<dim3(nXE / 256), dim3(256), 0, stream>>>(x, proj, nXE);
  }

  // final norm + LM head (dominant GEMM: 2048 x 32000 x 512)
  rms_kernel<<<dim3(kBT), dim3(256), 0, stream>>>(x, rms_f_w, xn);
  launch_gemm(stream, true, xn, kE, 0, 0, head_w, kE, 0, 0, out, kV, 0, 0, kBT, kV, kE, 1, 1);
}
